// HoyerSparsityLoss_73040213835966
// MI455X (gfx1250) — compile-verified, matrix operand order checked by reading, not run on hardware
//
#include <hip/hip_runtime.h>
#include <hip/hip_bf16.h>

typedef __attribute__((ext_vector_type(16))) _Float16 v16h;
typedef __attribute__((ext_vector_type(8)))  float    v8f;

#define N_ROWS 1024
#define DIM    256
#define ASTR   (DIM + 4)     // padded A row stride (floats): 260 -> bank shift 4/row
#define KC     64            // K-chunk staged to LDS per tile
#define BSTR   (KC + 4)      // padded B row stride: 68
#define NGROUPS 8            // column groups (128 tiles / 8 = 16 tiles per block)
#define TILES_PER_GROUP 16
#define INV_T   20.0f        // 1 / 0.05
#define SQRT_D  16.0f        // sqrt(256)
#define INV_SD1 (1.0f / 15.0f)

// -------- CDNA5 async global->LDS staging via inline asm (ASYNCcnt path) --------
#if defined(__gfx1250__)
#define HAVE_ASYNC 1
__device__ __forceinline__ void async_cp_b128(float* dst_lds, const float* src_glb) {
  // LDS byte offset: generic -> AS(3) -> 32-bit LDS address
  unsigned int loff =
      (unsigned int)(unsigned long long)(__attribute__((address_space(3))) float*)dst_lds;
  unsigned long long gaddr = (unsigned long long)src_glb;
  asm volatile("global_load_async_to_lds_b128 %0, %1, off"
               :
               : "v"(loff), "v"(gaddr)
               : "memory");
}
__device__ __forceinline__ void async_wait0() {
  asm volatile("s_wait_asynccnt 0" ::: "memory");
}
__device__ __forceinline__ void ds_fence0() {
  asm volatile("s_wait_dscnt 0" ::: "memory");
}
#else
#define HAVE_ASYNC 0
#endif

// ---------------- squared row norms for anchor / pos / neg ----------------
__global__ void hoyer_norms(const float* __restrict__ A, const float* __restrict__ P,
                            const float* __restrict__ G, float* __restrict__ norms) {
  int wave = (blockIdx.x * blockDim.x + threadIdx.x) >> 5;
  int lane = threadIdx.x & 31;
  const float* src; int row;
  if (wave < N_ROWS)          { src = A; row = wave; }
  else if (wave < 2 * N_ROWS) { src = P; row = wave - N_ROWS; }
  else                        { src = G; row = wave - 2 * N_ROWS; }
  const float4* r = (const float4*)(src + (size_t)row * DIM);
  float s = 0.f;
  for (int i = lane; i < DIM / 4; i += 32) {
    float4 v = r[i];
    s += v.x * v.x + v.y * v.y + v.z * v.z + v.w * v.w;
  }
  for (int off = 16; off > 0; off >>= 1) s += __shfl_xor(s, off, 32);
  if (lane == 0) norms[wave] = s;
}

// ---------------- main pairwise kernel ----------------
// grid: 64 row-blocks x 8 col-groups; block: 128 threads (4 wave32)
__launch_bounds__(128)
__global__ void hoyer_main(const float* __restrict__ anchor, const float* __restrict__ pos,
                           const float* __restrict__ neg, const float* __restrict__ norms,
                           float* __restrict__ ws_rowsum, float* __restrict__ ws_diag) {
  __shared__ __align__(16) float Alds[16 * ASTR];      // 16 anchor rows, f32
  __shared__ __align__(16) float Blds[4][16 * BSTR];   // per-wave B chunk
  __shared__ float lds_na[16];
  __shared__ float lds_diag[16];
  __shared__ float lds_wsum[4][16];

  const int tid  = threadIdx.x;
  const int wave = tid >> 5;
  const int lane = tid & 31;
  const int rb   = (int)blockIdx.x >> 3;   // anchor row block 0..63
  const int g    = (int)blockIdx.x & 7;    // column group 0..7
  const int row0 = rb * 16;

  // stage 16x256 anchor block into LDS
  {
    const float* aptr = anchor + (size_t)row0 * DIM;
#if HAVE_ASYNC
    for (int i = tid; i < 16 * (DIM / 4); i += 128) {
      int r = i >> 6, c = i & 63;
      async_cp_b128(&Alds[r * ASTR + c * 4], aptr + (size_t)i * 4);
    }
#else
    const float4* src = (const float4*)aptr;
    for (int i = tid; i < 16 * (DIM / 4); i += 128) {
      int r = i >> 6, c = i & 63;
      float4 v = src[i];
      float* dst = &Alds[r * ASTR + c * 4];
      dst[0] = v.x; dst[1] = v.y; dst[2] = v.z; dst[3] = v.w;
    }
#endif
    if (tid < 16) { lds_na[tid] = norms[row0 + tid]; lds_diag[tid] = 0.f; }
#if HAVE_ASYNC
    async_wait0();       // LDS writes visible before the barrier
#endif
  }
  __syncthreads();

  const int half  = lane >> 4;
  const int nloc  = lane & 15;     // output column within tile (C-fragment N)
  const int mbase = half * 8;      // output row base (C-fragment M group)

  float rowacc[8];
#pragma unroll
  for (int v = 0; v < 8; ++v) rowacc[v] = 0.f;

  float* Bl = &Blds[wave][0];

  for (int ctl = wave; ctl < TILES_PER_GROUP; ctl += 4) {
    const int ct = g * TILES_PER_GROUP + ctl;          // global tile 0..127
    const float* tsrc = (ct < 64) ? pos : neg;
    const int trow0 = (ct < 64) ? ct * 16 : (ct - 64) * 16;
    const float nb = norms[N_ROWS + ct * 16 + nloc];

    float accL1[8];
#pragma unroll
    for (int v = 0; v < 8; ++v) accL1[v] = 0.f;
    v8f cfrag = {};

    for (int kc = 0; kc < DIM / KC; ++kc) {
      // stage 16x64 target chunk into the wave-private LDS buffer
      const float* bsrc = tsrc + (size_t)trow0 * DIM + kc * KC;
#if HAVE_ASYNC
      // prior-chunk LDS reads must retire before async writes can land in Bl
      ds_fence0();
      for (int i = lane; i < 16 * (KC / 4); i += 32) {
        int r = i >> 4, c4 = i & 15;
        async_cp_b128(&Bl[r * BSTR + c4 * 4], bsrc + (size_t)r * DIM + c4 * 4);
      }
      async_wait0();
#else
      for (int i = lane; i < 16 * (KC / 4); i += 32) {
        int r = i >> 4, c4 = i & 15;
        float4 v = *(const float4*)(bsrc + (size_t)r * DIM + c4 * 4);
        float* dst = &Bl[r * BSTR + c4 * 4];
        dst[0] = v.x; dst[1] = v.y; dst[2] = v.z; dst[3] = v.w;
      }
#endif

      // dot-product part via WMMA: two K=32 steps, f16 in / f32 accumulate
#pragma unroll
      for (int ks = 0; ks < 2; ++ks) {
        const int kb = mbase;      // 16-bit A/B fragment K-base = 8*half
        v16h af, bf;
        const float* Ar = &Alds[nloc * ASTR + kc * KC + ks * 32 + kb];
        const float* Br = &Bl[nloc * BSTR + ks * 32 + kb];
#pragma unroll
        for (int e = 0; e < 8; ++e) {
          af[e]     = (_Float16)Ar[e];
          af[8 + e] = (_Float16)Ar[16 + e];
          bf[e]     = (_Float16)Br[e];
          bf[8 + e] = (_Float16)Br[16 + e];
        }
        cfrag = __builtin_amdgcn_wmma_f32_16x16x32_f16(false, af, false, bf,
                                                       (short)0, cfrag, false, false);
      }

      // L1 part (pure VALU): A reads are half-wave broadcasts, B reads conflict-free
      const float4* Brow = (const float4*)&Bl[nloc * BSTR];
#pragma unroll 4
      for (int d4 = 0; d4 < KC / 4; ++d4) {
        float4 bv = Brow[d4];
#pragma unroll
        for (int v = 0; v < 8; ++v) {
          const float4 av = *(const float4*)&Alds[(mbase + v) * ASTR + kc * KC + d4 * 4];
          accL1[v] += fabsf(av.x - bv.x) + fabsf(av.y - bv.y)
                    + fabsf(av.z - bv.z) + fabsf(av.w - bv.w);
        }
      }
    }

    // tile epilogue: hoyer + exp; C-fragment layout == accL1 layout
#pragma unroll
    for (int v = 0; v < 8; ++v) {
      const int m = mbase + v;
      const float dot  = cfrag[v];
      const float l2sq = lds_na[m] + nb - 2.f * dot;
      const float l2   = sqrtf(fmaxf(l2sq, 0.f)) + 1e-8f;
      const float hoyer = (SQRT_D - accL1[v] / l2) * INV_SD1;
      rowacc[v] += __expf(hoyer * INV_T);
      if (ct == rb && nloc == m) lds_diag[m] = hoyer;   // positive-pair diagonal
    }
  }

  // reduce exp-sums across the 16 lanes of each half-wave
#pragma unroll
  for (int v = 0; v < 8; ++v)
    for (int off = 1; off < 16; off <<= 1)
      rowacc[v] += __shfl_xor(rowacc[v], off, 32);
  if (nloc == 0) {
#pragma unroll
    for (int v = 0; v < 8; ++v) lds_wsum[wave][mbase + v] = rowacc[v];
  }
  __syncthreads();

  if (tid < 16) {
    float s = lds_wsum[0][tid] + lds_wsum[1][tid] + lds_wsum[2][tid] + lds_wsum[3][tid];
    ws_rowsum[((size_t)rb * NGROUPS + g) * 16 + tid] = s;
    if (g == (rb >> 4)) ws_diag[row0 + tid] = lds_diag[tid];  // diag tile lives in this group
  }
}

// ---------------- final log-sum-exp combine + mean ----------------
__global__ void hoyer_finalize(const float* __restrict__ ws_rowsum,
                               const float* __restrict__ ws_diag,
                               float* __restrict__ out) {
  __shared__ float red[32];
  const int i = threadIdx.x;            // 0..1023 = anchor row
  const int rb = i >> 4, m = i & 15;
  float s = 0.f;
#pragma unroll
  for (int g = 0; g < NGROUPS; ++g)
    s += ws_rowsum[((size_t)rb * NGROUPS + g) * 16 + m];
  float loss = __logf(s) - ws_diag[i] * INV_T;
  for (int off = 16; off > 0; off >>= 1) loss += __shfl_xor(loss, off, 32);
  if ((i & 31) == 0) red[i >> 5] = loss;
  __syncthreads();
  if (i < 32) {
    float v = red[i];
    for (int off = 16; off > 0; off >>= 1) v += __shfl_xor(v, off, 32);
    if (i == 0) out[0] = v * (1.0f / (float)N_ROWS);
  }
}

extern "C" void kernel_launch(void* const* d_in, const int* in_sizes, int n_in,
                              void* d_out, int out_size, void* d_ws, size_t ws_size,
                              hipStream_t stream) {
  (void)in_sizes; (void)n_in; (void)out_size; (void)ws_size;
  const float* anchor = (const float*)d_in[0];
  const float* pos    = (const float*)d_in[1];
  const float* neg    = (const float*)d_in[2];
  float* ws     = (float*)d_ws;
  float* norms  = ws;                    // 3072 floats
  float* rowsum = ws + 3072;             // 64*8*16 = 8192 floats
  float* diag   = ws + 3072 + 8192;      // 1024 floats

  hoyer_norms   <<<384, 256, 0, stream>>>(anchor, pos, neg, norms);
  hoyer_main    <<<64 * NGROUPS, 128, 0, stream>>>(anchor, pos, neg, norms, rowsum, diag);
  hoyer_finalize<<<1, 1024, 0, stream>>>(rowsum, diag, (float*)d_out);
}
